// Self_Attention_23304492548398
// MI455X (gfx1250) — compile-verified, matrix operand order checked by reading, not run on hardware
//
#include <hip/hip_runtime.h>

// ---------------- types ----------------
typedef __bf16 bf16_t;
typedef __attribute__((ext_vector_type(16))) __bf16 v16bf;
typedef __attribute__((ext_vector_type(8)))  __bf16 v8bf;
typedef __attribute__((ext_vector_type(4)))  __bf16 v4bf;
typedef __attribute__((ext_vector_type(8)))  float  v8f;
typedef __attribute__((ext_vector_type(4)))  float  v4f;

#define B_DIM  8
#define L_DIM  2048
#define D_DIM  1024

__device__ __forceinline__ bf16_t f2bf(float f) {
    // round-to-nearest-even f32 -> bf16
    unsigned u = __float_as_uint(f);
    unsigned r = (u + 0x7FFFu + ((u >> 16) & 1u)) >> 16;
    unsigned short s = (unsigned short)r;
    return __builtin_bit_cast(bf16_t, s);
}

__device__ __forceinline__ v8f vzero8() {
    v8f z;
#pragma unroll
    for (int i = 0; i < 8; ++i) z[i] = 0.0f;
    return z;
}

__device__ __forceinline__ v16bf combine16(v8bf lo, v8bf hi) {
    v16bf r;
#pragma unroll
    for (int i = 0; i < 8; ++i) { r[i] = lo[i]; r[8 + i] = hi[i]; }
    return r;
}

// lane holds K-segments [8*hi, 8*hi+8) and [16+8*hi, 16+8*hi+8)  (ISA 16-bit A/B layout)
__device__ __forceinline__ v16bf load_frag(const bf16_t* p) {
    v8bf lo = *(const v8bf*)p;
    v8bf hi = *(const v8bf*)(p + 16);
    return combine16(lo, hi);
}

__device__ __forceinline__ v8f wmma_bf16(v16bf a, v16bf b, v8f c) {
    return __builtin_amdgcn_wmma_f32_16x16x32_bf16(false, a, false, b, (short)0, c,
                                                   false, false);
}

// ---------------- prep: fp32 -> bf16 ----------------
__global__ __launch_bounds__(256) void k_cvt_x(const float* __restrict__ x,
                                               bf16_t* __restrict__ xb, int n) {
    int i = blockIdx.x * 256 + threadIdx.x;
    if (i < n) xb[i] = f2bf(x[i]);
}

// W [K=1024][N=1024] row-major  ->  Wt [N][K] bf16 (so B-fragments are contiguous)
__global__ __launch_bounds__(256) void k_wt(const float* __restrict__ W,
                                            bf16_t* __restrict__ Wt) {
    int i = blockIdx.x * 256 + threadIdx.x;      // 1M elements
    int k = i >> 10, n = i & 1023;
    Wt[(size_t)n * D_DIM + k] = f2bf(W[i]);
}

// ---------------- QKV projection GEMM ----------------
// out[m][n] = sum_k xb[m][k] * Wt[n][k] + bias[n]
// M = B*L = 16384, N = K = 1024. Block tile 128x128, 8 waves (4 in M x 2 in N),
// wave tile 32x64 = 2x4 accumulator 16x16 tiles.
__global__ __launch_bounds__(256) void k_qkv(const bf16_t* __restrict__ xb,
                                             const bf16_t* __restrict__ Wt,
                                             const float* __restrict__ bias,
                                             bf16_t* __restrict__ dst,
                                             int storeVT) {
    const int wave = threadIdx.x >> 5;
    const int lane = threadIdx.x & 31;
    const int nl = lane & 15;
    const int hi = lane >> 4;
    const int wm = wave & 3;
    const int wn = wave >> 2;
    const int row0 = blockIdx.x * 128 + wm * 32;
    const int col0 = blockIdx.y * 128 + wn * 64;

    v8f acc[2][4];
#pragma unroll
    for (int i = 0; i < 2; ++i)
#pragma unroll
        for (int j = 0; j < 4; ++j) acc[i][j] = vzero8();

    const bf16_t* arow0 = xb + (size_t)(row0 + nl) * D_DIM + 8 * hi;
    const bf16_t* arow1 = arow0 + (size_t)16 * D_DIM;

    for (int kk = 0; kk < D_DIM; kk += 32) {
        v16bf a0 = load_frag(arow0 + kk);
        v16bf a1 = load_frag(arow1 + kk);
#pragma unroll
        for (int j = 0; j < 4; ++j) {
            const bf16_t* bp = Wt + (size_t)(col0 + j * 16 + nl) * D_DIM + kk + 8 * hi;
            v16bf bb = load_frag(bp);
            acc[0][j] = wmma_bf16(a0, bb, acc[0][j]);
            acc[1][j] = wmma_bf16(a1, bb, acc[1][j]);
        }
    }

#pragma unroll
    for (int j = 0; j < 4; ++j) {
        int c = col0 + j * 16 + nl;
        float bv = bias[c];
#pragma unroll
        for (int i = 0; i < 2; ++i) {
#pragma unroll
            for (int r = 0; r < 8; ++r) {
                int m = row0 + i * 16 + 8 * hi + r;          // C/D layout: M = r + 8*hi
                float v = acc[i][j][r] + bv;
                if (!storeVT) {
                    dst[(size_t)m * D_DIM + c] = f2bf(v);    // row-major [M][1024]
                } else {
                    int bidx = m >> 11;                      // Vt [B][DV][L]
                    int l    = m & (L_DIM - 1);
                    dst[((size_t)bidx * D_DIM + c) * L_DIM + l] = f2bf(v);
                }
            }
        }
    }
}

// ---------------- attention (flash-style, padding-aware softmax) ----------------
// Block = (batch b, 16-query tile). 8 waves, 64-key chunks:
//   score GEMM: wave w covers DK slice [w*128, w*128+128) -> partials reduced in LDS
//   output GEMM: wave w covers DV slice [w*128, w*128+128) -> 8 acc tiles
// 2 barriers per 64 keys (32 chunks -> 64 barriers/block).
__global__ __launch_bounds__(256) void k_attn(const bf16_t* __restrict__ Qb,
                                              const bf16_t* __restrict__ Kb,
                                              const bf16_t* __restrict__ Vt,
                                              const int* __restrict__ actu,
                                              float* __restrict__ out) {
    __shared__ __align__(16) float Spart[8][16][64];       // 32 KB partial scores
    __shared__ __align__(16) float Sexp[16][64];           //  4 KB fp32 exp(scores)
    __shared__ __align__(16) bf16_t Pt[16][64];            //  2 KB bf16 P tile
    __shared__ float inv16[16];

    const int wave = threadIdx.x >> 5;
    const int lane = threadIdx.x & 31;
    const int nl = lane & 15;
    const int hi = lane >> 4;
    const int b = blockIdx.y;
    const int qt = blockIdx.x;
    const float scale = 0.03125f;                          // 1/sqrt(1024)

    // preload Q A-fragments for this wave's DK slice (invariant over key loop)
    const size_t qbase = ((size_t)(b * L_DIM + qt * 16 + nl)) * D_DIM + wave * 128 + 8 * hi;
    v16bf qf[4];
#pragma unroll
    for (int ks = 0; ks < 4; ++ks) qf[ks] = load_frag(Qb + qbase + ks * 32);

    v8f oacc[8];
#pragma unroll
    for (int t = 0; t < 8; ++t) oacc[t] = vzero8();
    float rowsum = 0.0f;                                   // live in wave 0, lanes 0..15

    for (int ch = 0; ch < L_DIM / 64; ++ch) {
        const int key0 = ch * 64;

        // ---- partial scores (16q x 64k) over this wave's DK slice ----
        v8f s[4];
#pragma unroll
        for (int kt = 0; kt < 4; ++kt) s[kt] = vzero8();
        const size_t kr = ((size_t)(b * L_DIM + key0 + nl)) * D_DIM + wave * 128 + 8 * hi;
#pragma unroll
        for (int ks = 0; ks < 4; ++ks) {
#pragma unroll
            for (int kt = 0; kt < 4; ++kt) {
                v16bf bb = load_frag(Kb + kr + (size_t)(kt * 16) * D_DIM + ks * 32);
                s[kt] = wmma_bf16(qf[ks], bb, s[kt]);
            }
        }
#pragma unroll
        for (int kt = 0; kt < 4; ++kt) {
#pragma unroll
            for (int r = 0; r < 8; ++r) {
                int m = 8 * hi + r;
                Spart[wave][m][kt * 16 + nl] = s[kt][r];
            }
        }
        __syncthreads();

        // ---- reduce partials, exp in fp32, write bf16 P (4 contiguous elems/thread) ----
        {
            int idx = threadIdx.x * 4;                     // 1024 elements = 16 x 64
            int m = idx >> 6, n = idx & 63;
            v4f v;
#pragma unroll
            for (int i = 0; i < 4; ++i) v[i] = 0.0f;
#pragma unroll
            for (int w = 0; w < 8; ++w) {
                v4f p = *(const v4f*)&Spart[w][m][n];
#pragma unroll
                for (int i = 0; i < 4; ++i) v[i] += p[i];
            }
            v4f e;
            v4bf pb;
#pragma unroll
            for (int i = 0; i < 4; ++i) {
                e[i] = __expf(v[i] * scale);
                pb[i] = f2bf(e[i]);
            }
            *(v4f*)&Sexp[m][n] = e;
            *(v4bf*)&Pt[m][n] = pb;
        }
        __syncthreads();

        // wave 0 keeps the fp32 running row-sum (matches reference tsum numerics)
        if (wave == 0 && lane < 16) {
#pragma unroll
            for (int n = 0; n < 64; n += 4) {
                v4f e = *(const v4f*)&Sexp[lane][n];
                rowsum += e[0] + e[1] + e[2] + e[3];
            }
        }

        // ---- O += P (16x64) x V (64 x 16) per DV tile, two K=32 steps ----
#pragma unroll
        for (int ksub = 0; ksub < 2; ++ksub) {
            v16bf pf = load_frag(&Pt[nl][ksub * 32 + 8 * hi]);
#pragma unroll
            for (int t = 0; t < 8; ++t) {
                int c = wave * 128 + t * 16 + nl;
                const bf16_t* vp = Vt + ((size_t)b * D_DIM + c) * L_DIM
                                 + key0 + ksub * 32 + 8 * hi;
                v16bf vf = load_frag(vp);
                oacc[t] = wmma_bf16(pf, vf, oacc[t]);
            }
        }
    }

    if (wave == 0 && lane < 16) {
        float pad = (float)(L_DIM - actu[b]);
        inv16[lane] = 1.0f / (rowsum - pad);
    }
    __syncthreads();

#pragma unroll
    for (int t = 0; t < 8; ++t) {
#pragma unroll
        for (int r = 0; r < 8; ++r) {
            int m = 8 * hi + r;
            int row = qt * 16 + m;
            int c = wave * 128 + t * 16 + nl;
            out[((size_t)b * L_DIM + row) * D_DIM + c] = oacc[t][r] * inv16[m];
        }
    }
}

// ---------------- host-side launcher ----------------
extern "C" void kernel_launch(void* const* d_in, const int* in_sizes, int n_in,
                              void* d_out, int out_size, void* d_ws, size_t ws_size,
                              hipStream_t stream) {
    const float* x    = (const float*)d_in[0];
    const int*   actu = (const int*)  d_in[1];
    const float* Wq   = (const float*)d_in[2];
    const float* bq   = (const float*)d_in[3];
    const float* Wk   = (const float*)d_in[4];
    const float* bk   = (const float*)d_in[5];
    const float* Wv   = (const float*)d_in[6];
    const float* bv   = (const float*)d_in[7];
    float* out = (float*)d_out;

    char* ws = (char*)d_ws;
    size_t off = 0;
    const size_t XN  = (size_t)B_DIM * L_DIM * D_DIM;      // 16,777,216
    const size_t WN  = (size_t)D_DIM * D_DIM;              // 1,048,576
    bf16_t* xb  = (bf16_t*)(ws + off); off += XN * 2;      // 32 MB
    bf16_t* Wtq = (bf16_t*)(ws + off); off += WN * 2;      //  2 MB
    bf16_t* Wtk = (bf16_t*)(ws + off); off += WN * 2;
    bf16_t* Wtv = (bf16_t*)(ws + off); off += WN * 2;
    bf16_t* Qb  = (bf16_t*)(ws + off); off += XN * 2;      // 32 MB
    bf16_t* Kb  = (bf16_t*)(ws + off); off += XN * 2;      // 32 MB
    bf16_t* Vt  = (bf16_t*)(ws + off); off += XN * 2;      // 32 MB (transposed [B][DV][L])

    // prep
    k_cvt_x<<<dim3((unsigned)((XN + 255) / 256)), dim3(256), 0, stream>>>(x, xb, (int)XN);
    k_wt<<<dim3((unsigned)(WN / 256)), dim3(256), 0, stream>>>(Wq, Wtq);
    k_wt<<<dim3((unsigned)(WN / 256)), dim3(256), 0, stream>>>(Wk, Wtk);
    k_wt<<<dim3((unsigned)(WN / 256)), dim3(256), 0, stream>>>(Wv, Wtv);

    // QKV projections (M=16384 rows, N=1024 cols; 128x128 block tiles)
    dim3 gA(128, 8);
    k_qkv<<<gA, dim3(256), 0, stream>>>(xb, Wtq, bq, Qb, 0);
    k_qkv<<<gA, dim3(256), 0, stream>>>(xb, Wtk, bk, Kb, 0);
    k_qkv<<<gA, dim3(256), 0, stream>>>(xb, Wtv, bv, Vt, 1);

    // attention: one block per (16-query tile, batch)
    dim3 gB(L_DIM / 16, B_DIM);
    k_attn<<<gB, dim3(256), 0, stream>>>(Qb, Kb, Vt, actu, out);
}
